// CS2GQAAttention_9242769622290
// MI455X (gfx1250) — matrix operand order repaired and checked by reading, not compiled, not run on hardware
//
#include <hip/hip_runtime.h>
#include <hip/hip_bf16.h>

// ---------------------------------------------------------------------------
// Problem constants (from the reference)
// ---------------------------------------------------------------------------
#define B_    4
#define L_    896
#define D_    2048
#define HQ_   32
#define HKV_  8
#define HD_   64
#define GROUP_ 4
#define TPF_  7            // tokens per frame
#define DKV_  (HKV_*HD_)   // 512
#define NCOL_ (D_ + 2*DKV_) // 3072 fused qkv columns

typedef __attribute__((ext_vector_type(16))) __bf16 v16bf;
typedef __attribute__((ext_vector_type(8)))  float  v8f;
typedef __attribute__((ext_vector_type(4)))  unsigned int v4u;
typedef __attribute__((ext_vector_type(2)))  float  v2f;

union Frag16 {
    v16bf v;
    v4u   q[2];
    unsigned short s[16];
};

// float -> bf16 bit pattern, round-to-nearest-even
__device__ __forceinline__ unsigned short f2bf(float f) {
    unsigned u = __float_as_uint(f);
    unsigned r = u + 0x7FFFu + ((u >> 16) & 1u);
    return (unsigned short)(r >> 16);
}

// A-matrix fragment (16 x 32, bf16, row-major source, K contiguous).
// Lanes 0-15: row = lane,    elems 0-7 = K k0..k0+7,   elems 8-15 = K k0+16..k0+23
// Lanes 16-31: row = lane-16, elems 0-7 = K k0+8..k0+15, elems 8-15 = K k0+24..k0+31
__device__ __forceinline__ Frag16 load_a_frag(const unsigned short* A, int lda,
                                              int row0, int k0) {
    int lane = threadIdx.x & 31;
    int half = lane >> 4;
    int row  = row0 + (lane & 15);
    const unsigned short* p = A + (size_t)row * lda + k0 + half * 8;
    Frag16 f;
    f.q[0] = *reinterpret_cast<const v4u*>(p);
    f.q[1] = *reinterpret_cast<const v4u*>(p + 16);
    return f;
}

// B-matrix fragment (32 x 16, bf16, row-major source, N contiguous).
// Column = n0 + lane%16; elems 0-7 = K k0+8h..k0+8h+7, elems 8-15 = K k0+16+8h..k0+23+8h (h = lane/16).
__device__ __forceinline__ Frag16 load_b_frag(const unsigned short* Bm, int ldb,
                                              int k0, int n0) {
    int lane = threadIdx.x & 31;
    const unsigned short* p = Bm + (size_t)(k0 + (lane >> 4) * 8) * ldb + n0 + (lane & 15);
    Frag16 f;
#pragma unroll
    for (int s = 0; s < 8; ++s) {
        f.s[s]     = p[(size_t)s * ldb];
        f.s[s + 8] = p[(size_t)(s + 16) * ldb];
    }
    return f;
}

__device__ __forceinline__ v8f zero_v8f() {
    v8f z;
#pragma unroll
    for (int i = 0; i < 8; ++i) z[i] = 0.0f;
    return z;
}

// ---------------------------------------------------------------------------
// Kernel 0: fp32 -> bf16 bulk conversion
// ---------------------------------------------------------------------------
__global__ void cvt_f32_bf16(const float* __restrict__ src,
                             unsigned short* __restrict__ dst, int n) {
    int i = blockIdx.x * blockDim.x + threadIdx.x;
    if (i < n) dst[i] = f2bf(src[i]);
}

// ---------------------------------------------------------------------------
// Kernel 0b: RoPE cos/sin table, tab[(pos*32+i)] = {cos, sin}
// ---------------------------------------------------------------------------
__global__ void rope_table_kernel(const int* __restrict__ pos_ids,
                                  float* __restrict__ tab) {
    int idx = blockIdx.x * blockDim.x + threadIdx.x;   // over L_*32
    if (idx >= L_ * 32) return;
    int pos = idx >> 5;
    int i   = idx & 31;
    float pid  = (float)pos_ids[pos];
    float invf = __expf(-(float)i * (9.210340371976184f / 32.0f));
    float fr = pid * invf;
    float sn, cs;
    __sincosf(fr, &sn, &cs);
    tab[idx * 2]     = cs;
    tab[idx * 2 + 1] = sn;
}

// ---------------------------------------------------------------------------
// Kernel 1: fused QKV projection + RoPE epilogue.
// Each wave computes a 32x64 tile of the fused 3584 x 3072 GEMM
// (2 A-fragments reuse each B-fragment; 8 WMMAs per K-step).
// ---------------------------------------------------------------------------
__global__ __launch_bounds__(256) void qkv_rope_kernel(
    const unsigned short* __restrict__ xb,
    const unsigned short* __restrict__ wqb,
    const unsigned short* __restrict__ wkb,
    const unsigned short* __restrict__ wvb,
    const float* __restrict__ rope_tab,
    unsigned short* __restrict__ qb,    // [B][HQ][L][HD]
    unsigned short* __restrict__ kTb,   // [B][HKV][HD][L]  (transposed!)
    unsigned short* __restrict__ vb) {  // [B][HKV][L][HD]
    const int NG = NCOL_ / 64;          // 48 column groups
    int wave  = threadIdx.x >> 5;
    int lane  = threadIdx.x & 31;
    int wtile = blockIdx.x * 8 + wave;
    int mt = wtile / NG;
    int ng = wtile % NG;
    int m0 = mt * 32;
    int n0 = ng * 64;

    v8f acc[2][4];
#pragma unroll
    for (int a = 0; a < 2; ++a)
#pragma unroll
        for (int f = 0; f < 4; ++f) acc[a][f] = zero_v8f();

    for (int k0 = 0; k0 < D_; k0 += 32) {
        Frag16 a0 = load_a_frag(xb, D_, m0,      k0);
        Frag16 a1 = load_a_frag(xb, D_, m0 + 16, k0);
        if (k0 + 32 < D_) {   // hint next A rows into cache
            __builtin_prefetch(xb + (size_t)(m0 + (lane & 15)) * D_ + k0 + 32, 0, 1);
        }
#pragma unroll
        for (int f = 0; f < 4; ++f) {
            int col = n0 + f * 16;          // uniform across the wave
            const unsigned short* Bm;
            int ldb, bcol;
            if (col < D_)              { Bm = wqb; ldb = D_;   bcol = col; }
            else if (col < D_ + DKV_)  { Bm = wkb; ldb = DKV_; bcol = col - D_; }
            else                       { Bm = wvb; ldb = DKV_; bcol = col - (D_ + DKV_); }
            Frag16 b = load_b_frag(Bm, ldb, k0, bcol);
            acc[0][f] = __builtin_amdgcn_wmma_f32_16x16x32_bf16(
                false, a0.v, false, b.v, (short)0, acc[0][f], false, false);
            acc[1][f] = __builtin_amdgcn_wmma_f32_16x16x32_bf16(
                false, a1.v, false, b.v, (short)0, acc[1][f], false, false);
        }
    }

    // Epilogue: table-based RoPE for q/k columns, then scatter-store bf16.
    int half = lane >> 4;
    int nl   = lane & 15;
#pragma unroll
    for (int a = 0; a < 2; ++a) {
#pragma unroll
        for (int f = 0; f < 4; ++f) {
            int colbase = n0 + f * 16;      // region uniform per f (64-aligned regions)
            bool isq = colbase < D_;
            bool isv = colbase >= D_ + DKV_;
#pragma unroll
            for (int r = 0; r < 8; ++r) {
                float val = acc[a][f][r];
                // full-EXEC pair exchange (even/odd head dims are adjacent lanes)
                float other = __shfl_xor(val, 1, 32);
                int col = colbase + nl;
                int t   = m0 + a * 16 + r + half * 8;  // flattened (b, pos)
                int b   = t / L_;
                int pos = t - b * L_;
                int c   = col & 63;                     // dim within head
                if (!isv) {                             // RoPE for q and k
                    int i = c >> 1;
                    v2f cssn = *reinterpret_cast<const v2f*>(
                        rope_tab + (size_t)(pos * 32 + i) * 2);
                    float term = other * cssn[1];
                    val = (c & 1) ? fmaf(val, cssn[0], term)
                                  : fmaf(val, cssn[0], -term);
                }
                unsigned short bv = f2bf(val);
                if (isq) {
                    int h = col >> 6;
                    qb[(((size_t)b * HQ_ + h) * L_ + pos) * HD_ + c] = bv;
                } else if (!isv) {
                    int hk = (col - D_) >> 6;
                    kTb[(((size_t)b * HKV_ + hk) * HD_ + c) * L_ + pos] = bv;
                } else {
                    int hk = (col - (D_ + DKV_)) >> 6;
                    vb[(((size_t)b * HKV_ + hk) * L_ + pos) * HD_ + c] = bv;
                }
            }
        }
    }
}

// ---------------------------------------------------------------------------
// Kernel 2: flash-style GQA attention with frame-causal mask.
// One wave per (b, h, 16-query tile). Keys processed in tiles of 32.
// ---------------------------------------------------------------------------
__global__ __launch_bounds__(256) void attn_kernel(
    const unsigned short* __restrict__ qb,
    const unsigned short* __restrict__ kTb,
    const unsigned short* __restrict__ vb,
    unsigned short* __restrict__ ob) {      // [B][L][HQ*HD]
    __shared__ __attribute__((aligned(16))) unsigned short Pl[8][16][32];

    int wave = threadIdx.x >> 5;
    int lane = threadIdx.x & 31;
    int half = lane >> 4;
    int nl   = lane & 15;

    const int QT = L_ / 16;                 // 56 query tiles
    int wtile = blockIdx.x * 8 + wave;
    int bh = wtile / QT;
    int qt = wtile % QT;
    int b  = bh / HQ_;
    int h  = bh % HQ_;
    int hk = h / GROUP_;
    int q0 = qt * 16;

    const unsigned short* Qbase = qb  + ((size_t)(b * HQ_  + h )) * L_ * HD_;
    const unsigned short* Kbase = kTb + ((size_t)(b * HKV_ + hk)) * HD_ * L_;
    const unsigned short* Vbase = vb  + ((size_t)(b * HKV_ + hk)) * L_ * HD_;

    // Q fragments for head-dim chunks 0..31 and 32..63
    Frag16 Qa  = load_a_frag(Qbase, HD_, q0, 0);
    Frag16 Qb2 = load_a_frag(Qbase, HD_, q0, 32);

    v8f O[4];
    float m[8], rs[8];
#pragma unroll
    for (int f = 0; f < 4; ++f) O[f] = zero_v8f();
#pragma unroll
    for (int r = 0; r < 8; ++r) { m[r] = -1e30f; rs[r] = 0.0f; }

    int maxA_tile = ((q0 + 15) / TPF_) * TPF_ + (TPF_ - 1); // max key any row needs
    int nk = (maxA_tile + 1 + 31) / 32;

    const float scale = 0.125f;             // 1/sqrt(64)

    for (int kt = 0; kt < nk; ++kt) {
        int kp0 = kt * 32;
        // ---- scores: S[16x32] = Q[16x64] * K^T[64x32] ----
        Frag16 b0 = load_b_frag(Kbase, L_, 0,  kp0);
        Frag16 b1 = load_b_frag(Kbase, L_, 32, kp0);
        v8f Sa = zero_v8f();
        Sa = __builtin_amdgcn_wmma_f32_16x16x32_bf16(false, Qa.v,  false, b0.v, (short)0, Sa, false, false);
        Sa = __builtin_amdgcn_wmma_f32_16x16x32_bf16(false, Qb2.v, false, b1.v, (short)0, Sa, false, false);
        Frag16 b2 = load_b_frag(Kbase, L_, 0,  kp0 + 16);
        Frag16 b3 = load_b_frag(Kbase, L_, 32, kp0 + 16);
        v8f Sb = zero_v8f();
        Sb = __builtin_amdgcn_wmma_f32_16x16x32_bf16(false, Qa.v,  false, b2.v, (short)0, Sb, false, false);
        Sb = __builtin_amdgcn_wmma_f32_16x16x32_bf16(false, Qb2.v, false, b3.v, (short)0, Sb, false, false);

        // ---- mask + online softmax (row stats live per D-fragment element) ----
#pragma unroll
        for (int r = 0; r < 8; ++r) {
            int qrow = q0 + r + half * 8;
            int maxk = (qrow / TPF_) * TPF_ + (TPF_ - 1);
            int ka = kp0 + nl;
            int kb = ka + 16;
            float sa = (ka <= maxk) ? Sa[r] * scale : -1e30f;
            float sb = (kb <= maxk) ? Sb[r] * scale : -1e30f;
            float tmax = fmaxf(sa, sb);
#pragma unroll
            for (int msk = 1; msk < 16; msk <<= 1)
                tmax = fmaxf(tmax, __shfl_xor(tmax, msk, 32));
            float newm  = fmaxf(m[r], tmax);
            float alpha = __expf(m[r] - newm);
            float pa = __expf(sa - newm);
            float pb = __expf(sb - newm);
            float psum = pa + pb;
#pragma unroll
            for (int msk = 1; msk < 16; msk <<= 1)
                psum += __shfl_xor(psum, msk, 32);
            rs[r] = rs[r] * alpha + psum;
            m[r] = newm;
#pragma unroll
            for (int f = 0; f < 4; ++f) O[f][r] *= alpha;
            int prow = r + half * 8;
            Pl[wave][prow][nl]      = f2bf(pa);
            Pl[wave][prow][nl + 16] = f2bf(pb);
        }

        // ---- P[16x32] * V[32x64]  (P restaged via per-wave LDS tile) ----
        Frag16 pf = load_a_frag(&Pl[wave][0][0], 32, 0, 0);
#pragma unroll
        for (int f = 0; f < 4; ++f) {
            Frag16 vf = load_b_frag(Vbase, HD_, kp0, f * 16);
            O[f] = __builtin_amdgcn_wmma_f32_16x16x32_bf16(
                false, pf.v, false, vf.v, (short)0, O[f], false, false);
        }
    }

    // ---- normalize + store bf16 into [B][L][2048] layout for O-proj ----
#pragma unroll
    for (int f = 0; f < 4; ++f) {
#pragma unroll
        for (int r = 0; r < 8; ++r) {
            int qrow = q0 + r + half * 8;
            float val = O[f][r] / rs[r];
            int col = h * HD_ + f * 16 + nl;
            ob[((size_t)b * L_ + qrow) * D_ + col] = f2bf(val);
        }
    }
}

// ---------------------------------------------------------------------------
// Kernel 3: output projection (3584 x 2048) @ (2048 x 2048) -> fp32 d_out
// Each wave computes a 32x64 tile (8 WMMAs per K-step).
// ---------------------------------------------------------------------------
__global__ __launch_bounds__(256) void oproj_kernel(
    const unsigned short* __restrict__ ob,
    const unsigned short* __restrict__ wob,
    float* __restrict__ out) {
    const int NG = D_ / 64;                 // 32 column groups
    int wave  = threadIdx.x >> 5;
    int lane  = threadIdx.x & 31;
    int wtile = blockIdx.x * 8 + wave;
    int mt = wtile / NG;
    int ng = wtile % NG;
    int m0 = mt * 32;
    int n0 = ng * 64;

    v8f acc[2][4];
#pragma unroll
    for (int a = 0; a < 2; ++a)
#pragma unroll
        for (int f = 0; f < 4; ++f) acc[a][f] = zero_v8f();

    for (int k0 = 0; k0 < D_; k0 += 32) {
        Frag16 a0 = load_a_frag(ob, D_, m0,      k0);
        Frag16 a1 = load_a_frag(ob, D_, m0 + 16, k0);
        if (k0 + 32 < D_) {
            __builtin_prefetch(ob + (size_t)(m0 + (lane & 15)) * D_ + k0 + 32, 0, 1);
        }
#pragma unroll
        for (int f = 0; f < 4; ++f) {
            Frag16 bf = load_b_frag(wob, D_, k0, n0 + f * 16);
            acc[0][f] = __builtin_amdgcn_wmma_f32_16x16x32_bf16(
                false, a0.v, false, bf.v, (short)0, acc[0][f], false, false);
            acc[1][f] = __builtin_amdgcn_wmma_f32_16x16x32_bf16(
                false, a1.v, false, bf.v, (short)0, acc[1][f], false, false);
        }
    }

    int half = lane >> 4;
    int nl   = lane & 15;
#pragma unroll
    for (int a = 0; a < 2; ++a) {
#pragma unroll
        for (int f = 0; f < 4; ++f) {
#pragma unroll
            for (int r = 0; r < 8; ++r) {
                int row = m0 + a * 16 + r + half * 8;
                out[(size_t)row * D_ + n0 + f * 16 + nl] = acc[a][f][r];
            }
        }
    }
}

// ---------------------------------------------------------------------------
// Launcher
// ---------------------------------------------------------------------------
extern "C" void kernel_launch(void* const* d_in, const int* in_sizes, int n_in,
                              void* d_out, int out_size, void* d_ws, size_t ws_size,
                              hipStream_t stream) {
    const float* x  = (const float*)d_in[0];
    const float* wq = (const float*)d_in[1];
    const float* wk = (const float*)d_in[2];
    const float* wv = (const float*)d_in[3];
    const float* wo = (const float*)d_in[4];
    const int* pos_ids = (const int*)d_in[5];
    float* out = (float*)d_out;

    const size_t NX  = (size_t)B_ * L_ * D_;         // 7,340,032
    const size_t NWQ = (size_t)D_ * D_;              // 4,194,304
    const size_t NWK = (size_t)D_ * DKV_;            // 1,048,576
    const size_t NKV = (size_t)B_ * HKV_ * HD_ * L_; // 1,835,008

    char* ws = (char*)d_ws;
    size_t off = 0;
    auto alloc_bytes = [&](size_t bytes) -> void* {
        void* p = ws + off;
        off += (bytes + 255) & ~(size_t)255;
        return p;
    };
    unsigned short* xb  = (unsigned short*)alloc_bytes(NX  * 2);
    unsigned short* wqb = (unsigned short*)alloc_bytes(NWQ * 2);
    unsigned short* wkb = (unsigned short*)alloc_bytes(NWK * 2);
    unsigned short* wvb = (unsigned short*)alloc_bytes(NWK * 2);
    unsigned short* wob = (unsigned short*)alloc_bytes(NWQ * 2);
    unsigned short* qb  = (unsigned short*)alloc_bytes(NX  * 2); // [B][HQ][L][HD]
    unsigned short* kTb = (unsigned short*)alloc_bytes(NKV * 2); // [B][HKV][HD][L]
    unsigned short* vb  = (unsigned short*)alloc_bytes(NKV * 2); // [B][HKV][L][HD]
    unsigned short* obuf= (unsigned short*)alloc_bytes(NX  * 2); // [B][L][2048]
    float* rope_tab     = (float*)alloc_bytes((size_t)L_ * 32 * 2 * sizeof(float));

    auto cvt = [&](const float* s, unsigned short* d, size_t n) {
        int ni = (int)n;
        cvt_f32_bf16<<<(ni + 255) / 256, 256, 0, stream>>>(s, d, ni);
    };
    cvt(x,  xb,  NX);
    cvt(wq, wqb, NWQ);
    cvt(wk, wkb, NWK);
    cvt(wv, wvb, NWK);
    cvt(wo, wob, NWQ);

    // RoPE cos/sin table: L * 32 entries
    rope_table_kernel<<<(L_ * 32 + 255) / 256, 256, 0, stream>>>(pos_ids, rope_tab);

    // QKV projection + RoPE: 112 row tiles * 48 col groups = 5376 wave tiles
    qkv_rope_kernel<<<5376 / 8, 256, 0, stream>>>(xb, wqb, wkb, wvb, rope_tab,
                                                  qb, kTb, vb);

    // Attention: 4 * 32 * 56 = 7168 wave tiles
    attn_kernel<<<7168 / 8, 256, 0, stream>>>(qb, kTb, vb, obuf);

    // Output projection: 112 * 32 = 3584 wave tiles
    oproj_kernel<<<3584 / 8, 256, 0, stream>>>(obuf, wob, out);
}
